// XuDecoder_38208029065709
// MI455X (gfx1250) — matrix operand order repaired and checked by reading, not compiled, not run on hardware
//
#include <hip/hip_runtime.h>
#include <hip/hip_bf16.h>

// Problem constants (from reference): T,B,V,E,H,C,S
#define T_ 32
#define B_ 256
#define V_ 10000
#define E_ 512
#define H_ 512
#define C_ 512
#define S_ 196
#define G3_ 1536   // 3*H

typedef __attribute__((ext_vector_type(16))) __bf16 v16bf;
typedef __attribute__((ext_vector_type(8)))  __bf16 v8bf;
typedef __attribute__((ext_vector_type(4)))  __bf16 v4bf;
typedef __attribute__((ext_vector_type(4)))  float  v4f;
typedef __attribute__((ext_vector_type(8)))  float  v8f;

static __device__ __forceinline__ __bf16 f2bf(float f) {
  unsigned u = __builtin_bit_cast(unsigned, f);
  unsigned r = u + 0x7FFFu + ((u >> 16) & 1u);   // round-to-nearest-even
  unsigned short h = (unsigned short)(r >> 16);
  return __builtin_bit_cast(__bf16, h);
}

static __device__ __forceinline__ float sigmoidf_(float x) {
  return 1.0f / (1.0f + __expf(-x));
}

// ---------------------------------------------------------------------------
// WMMA fragment mappings (v_wmma_f32_16x16x32_bf16, wave32):
// A (16x32 MxK): lane L -> M=L%16, half=L/16; elems 0..7 -> K=half*8+e,
//                elems 8..15 -> K=16+half*8+(e-8)  (two 8-long K runs).
// B (32x16 KxN): lane L -> N=L%16, half=L/16; elems 0..7 -> K=half*8+e, elems 8..15 -> K=16+half*8+(e-8) (same as A).
// C/D (16x16 f32): VGPR r -> M=r (lanes 0-15) / r+8 (lanes 16-31), N=L%16.
//
// Packed-B layout (built once per launch by pack_b_kernel): for each
// (n_tile, k_slab) a 512-element bf16 slab:
//   slab[chunk*256 + lane*8 + e7]  with  e = chunk*8+e7
// so a wave's fragment is two coalesced 16B-per-lane b128 loads.
// ---------------------------------------------------------------------------

__global__ void pack_b_kernel(const float* __restrict__ W, int N, int K,
                              __bf16* __restrict__ dst) {
  long idx = (long)blockIdx.x * 256 + threadIdx.x;   // K*N threads
  int  inslab = (int)(idx & 511);
  long slab   = idx >> 9;                            // tile_n * KS + ks
  int  KS     = K >> 5;
  int  tile_n = (int)(slab / KS);
  int  ks     = (int)(slab - (long)tile_n * KS);
  int  chunk  = inslab >> 8;
  int  lane   = (inslab >> 3) & 31;
  int  e      = (inslab & 7) + chunk * 8;            // 0..15
  int  n      = tile_n * 16 + (lane & 15);
  int  k      = ks * 32 + (lane >> 4) * 8 + e + chunk * 8;
  dst[idx] = f2bf(W[(long)k * N + n]);
}

// bulk f32 -> bf16 convert (4 elements/thread, vectorized)
__global__ void cvt_bf16_kernel(const float* __restrict__ src,
                                __bf16* __restrict__ dst, long n4) {
  long i = (long)blockIdx.x * 256 + threadIdx.x;
  if (i >= n4) return;
  v4f x = *(const v4f*)(src + i * 4);
  v4bf o;
#pragma unroll
  for (int j = 0; j < 4; ++j) o[j] = f2bf(x[j]);
  *(v4bf*)(dst + i * 4) = o;
}

// ---------------------------------------------------------------------------
// GEMM: D = A(MxK) @ Bpack(KxN) [+ D_prev] [+ bias];  A bf16 row-major.
// Each wave owns a 64-row x 16-col strip: 4 m-tiles share each B fragment,
// so every 32-K slab costs 2 B-loads + 8 A-loads for 4 independent WMMAs.
// flags bit0: accumulate into existing D.
// Grid: x = ceil(N/16/4), y = M/64, block = 128 (4 waves).
// K % 32 == 0, M % 64 == 0.
// ---------------------------------------------------------------------------
__global__ void wmma_gemm_kernel(const __bf16* __restrict__ A, int lda,
                                 const __bf16* __restrict__ Bp,
                                 float* __restrict__ D, int ldd,
                                 int M, int N, int K,
                                 const float* __restrict__ bias, int flags) {
  int lane   = threadIdx.x & 31;
  int wave   = threadIdx.x >> 5;
  int tile_n = blockIdx.x * 4 + wave;
  if (tile_n * 16 >= N) return;          // uniform per-wave exit; EXEC stays all-1
  int row0 = blockIdx.y * 64;
  int col0 = tile_n * 16;
  int half = lane >> 4;
  int KS   = K >> 5;

  v8f c[4];
  if (flags & 1) {
#pragma unroll
    for (int mt = 0; mt < 4; ++mt)
#pragma unroll
      for (int r = 0; r < 8; ++r) {
        int m = row0 + mt * 16 + ((lane < 16) ? r : r + 8);
        c[mt][r] = D[(long)m * ldd + col0 + (lane & 15)];
      }
  } else {
#pragma unroll
    for (int mt = 0; mt < 4; ++mt)
#pragma unroll
      for (int r = 0; r < 8; ++r) c[mt][r] = 0.0f;
  }

  const __bf16* slabp = Bp + ((long)tile_n * KS) * 512 + lane * 8;
  int m_a = row0 + (lane & 15);

  for (int ks = 0; ks < KS; ++ks) {
    union { v16bf v; v8bf h[2]; } b;
    b.h[0] = *(const v8bf*)(slabp);
    b.h[1] = *(const v8bf*)(slabp + 256);
    if (ks + 1 < KS) __builtin_prefetch(slabp + 512, 0, 1);
#pragma unroll
    for (int mt = 0; mt < 4; ++mt) {
      union { v16bf v; v8bf h[2]; } a;
      const __bf16* ap = A + (long)(m_a + mt * 16) * lda + ks * 32 + half * 8;
      a.h[0] = *(const v8bf*)(ap);
      a.h[1] = *(const v8bf*)(ap + 16);
      c[mt] = __builtin_amdgcn_wmma_f32_16x16x32_bf16(false, a.v, false, b.v,
                                                      (short)0, c[mt], false, false);
    }
    slabp += 512;
  }

  int n = col0 + (lane & 15);
  float bv = bias ? bias[n] : 0.0f;
#pragma unroll
  for (int mt = 0; mt < 4; ++mt)
#pragma unroll
    for (int r = 0; r < 8; ++r) {
      int m = row0 + mt * 16 + ((lane < 16) ? r : r + 8);
      D[(long)m * ldd + n] = c[mt][r] + bv;
    }
}

// ---------------------------------------------------------------------------
// Elementwise / reduction kernels
// ---------------------------------------------------------------------------
__global__ void zero_h_kernel(float* __restrict__ h, __bf16* __restrict__ hbf, int n) {
  int i = blockIdx.x * blockDim.x + threadIdx.x;
  if (i < n) { h[i] = 0.0f; hbf[i] = f2bf(0.0f); }
}

// y_emb[t,b,e] = emb_W[y[t,b], e]
__global__ void embed_kernel(const int* __restrict__ y, const float* __restrict__ embW,
                             float* __restrict__ yemb) {
  int idx = blockIdx.x * blockDim.x + threadIdx.x;     // T*B*E
  int e  = idx & (E_ - 1);
  int tb = idx >> 9;
  int tok = y[tb];
  yemb[idx] = embW[(long)tok * E_ + e];
}

// scores[s,b] = sum_c tanh(ctx_p[s,b,c] + hproj[b,c]) * mlpW[c] + mlp_b
__global__ void scores_kernel(const float* __restrict__ ctx_p,
                              const float* __restrict__ hproj,
                              const float* __restrict__ mlpW,
                              const float* __restrict__ mlpb,
                              float* __restrict__ scores) {
  int sb = blockIdx.x;                                  // s*B + b
  int b  = sb & (B_ - 1);
  const float* cp = ctx_p + (long)sb * C_;
  const float* hp = hproj + (long)b * C_;
  float acc = 0.0f;
  for (int c = threadIdx.x; c < C_; c += blockDim.x)
    acc += tanhf(cp[c] + hp[c]) * mlpW[c];
  __shared__ float red[256];
  red[threadIdx.x] = acc;
  __syncthreads();
  for (int o = 128; o > 0; o >>= 1) {
    if (threadIdx.x < o) red[threadIdx.x] += red[threadIdx.x + o];
    __syncthreads();
  }
  if (threadIdx.x == 0) scores[sb] = red[0] + mlpb[0];
}

// alpha[s,b] = softmax over s of scores[s,b]   (one block per b)
__global__ void softmax_alpha_kernel(const float* __restrict__ scores,
                                     float* __restrict__ alpha) {
  int b = blockIdx.x;
  int s = threadIdx.x;
  float v = (s < S_) ? scores[s * B_ + b] : -3.4e38f;
  __shared__ float red[256];
  red[s] = v;
  __syncthreads();
  for (int o = 128; o > 0; o >>= 1) {
    if (s < o) red[s] = fmaxf(red[s], red[s + o]);
    __syncthreads();
  }
  float mx = red[0];
  __syncthreads();
  float e = (s < S_) ? __expf(v - mx) : 0.0f;
  red[s] = e;
  __syncthreads();
  for (int o = 128; o > 0; o >>= 1) {
    if (s < o) red[s] += red[s + o];
    __syncthreads();
  }
  float inv = 1.0f / red[0];
  if (s < S_) alpha[s * B_ + b] = e * inv;
}

// xcat_bf[b, 0:512]    = bf16(y_emb[t,b,:])
// xcat_bf[b, 512:1024] = bf16(sum_s alpha[s,b] * ctx[s,b,:])
__global__ void zxcat_kernel(const float* __restrict__ alpha,
                             const float* __restrict__ ctx,
                             const float* __restrict__ yemb_t,
                             __bf16* __restrict__ xcat_bf) {
  int idx = blockIdx.x * blockDim.x + threadIdx.x;     // B*C
  int c = idx & (C_ - 1);
  int b = idx >> 9;
  float acc = 0.0f;
  for (int s = 0; s < S_; ++s)
    acc += alpha[s * B_ + b] * ctx[((long)(s * B_ + b)) * C_ + c];
  xcat_bf[(long)b * 1024 + 512 + c] = f2bf(acc);
  xcat_bf[(long)b * 1024 + c]       = f2bf(yemb_t[(long)b * E_ + c]);
}

// GRU gating: h_new = (1-z)*n + z*h    (writes f32 and bf16 mirrors)
__global__ void gru_kernel(const float* __restrict__ gi, const float* __restrict__ gh,
                           const float* __restrict__ h,
                           float* __restrict__ h_new, __bf16* __restrict__ h_new_bf) {
  int idx = blockIdx.x * blockDim.x + threadIdx.x;     // B*H
  int j = idx & (H_ - 1);
  int b = idx >> 9;
  const float* gib = gi + (long)b * G3_;
  const float* ghb = gh + (long)b * G3_;
  float ir = gib[j], iz = gib[H_ + j], in_ = gib[2 * H_ + j];
  float hr = ghb[j], hz = ghb[H_ + j], hn  = ghb[2 * H_ + j];
  float r  = sigmoidf_(ir + hr);
  float zz = sigmoidf_(iz + hz);
  float n  = tanhf(in_ + r * hn);
  float hv = (1.0f - zz) * n + zz * h[idx];
  h_new[idx]    = hv;
  h_new_bf[idx] = f2bf(hv);
}

// logit_bf = bf16(tanh(logit + y_emb_t))
__global__ void logit_finish_kernel(const float* __restrict__ logit,
                                    const float* __restrict__ yemb_t,
                                    __bf16* __restrict__ logit_bf) {
  int idx = blockIdx.x * blockDim.x + threadIdx.x;     // B*E
  logit_bf[idx] = f2bf(tanhf(logit[idx] + yemb_t[idx]));
}

// log-softmax over V per row b; write logps; per-(t,b) masked NLL -> loss_part
__global__ void logsoftmax_loss_kernel(const float* __restrict__ logits2,
                                       const int* __restrict__ y, int t,
                                       float* __restrict__ out_logp,
                                       float* __restrict__ loss_part) {
  int b = blockIdx.x;
  const float* row = logits2 + (long)b * V_;
  __shared__ float red[256];
  float mx = -3.4e38f;
  for (int v = threadIdx.x; v < V_; v += 256) mx = fmaxf(mx, row[v]);
  red[threadIdx.x] = mx;
  __syncthreads();
  for (int o = 128; o > 0; o >>= 1) {
    if (threadIdx.x < o) red[threadIdx.x] = fmaxf(red[threadIdx.x], red[threadIdx.x + o]);
    __syncthreads();
  }
  mx = red[0];
  __syncthreads();
  float s = 0.0f;
  for (int v = threadIdx.x; v < V_; v += 256) s += __expf(row[v] - mx);
  red[threadIdx.x] = s;
  __syncthreads();
  for (int o = 128; o > 0; o >>= 1) {
    if (threadIdx.x < o) red[threadIdx.x] += red[threadIdx.x + o];
    __syncthreads();
  }
  float logZ = mx + __logf(red[0]);
  float* op = out_logp + ((long)t * B_ + b) * V_;
  for (int v = threadIdx.x; v < V_; v += 256) op[v] = row[v] - logZ;
  if (threadIdx.x == 0) {
    int yn = y[(t + 1) * B_ + b];
    loss_part[t * B_ + b] = (yn != 0) ? -(row[yn] - logZ) : 0.0f;
  }
}

// deterministic fixed-order sum of (T-1)*B loss parts -> out[0]
__global__ void final_loss_kernel(const float* __restrict__ loss_part,
                                  float* __restrict__ out) {
  __shared__ float red[256];
  float acc = 0.0f;
  for (int i = threadIdx.x; i < (T_ - 1) * B_; i += 256) acc += loss_part[i];
  red[threadIdx.x] = acc;
  __syncthreads();
  for (int o = 128; o > 0; o >>= 1) {
    if (threadIdx.x < o) red[threadIdx.x] += red[threadIdx.x + o];
    __syncthreads();
  }
  if (threadIdx.x == 0) out[0] = red[0];
}

// ---------------------------------------------------------------------------
extern "C" void kernel_launch(void* const* d_in, const int* in_sizes, int n_in,
                              void* d_out, int out_size, void* d_ws, size_t ws_size,
                              hipStream_t stream) {
  (void)in_sizes; (void)n_in; (void)out_size; (void)ws_size;
  const int*   y         = (const int*)  d_in[0];
  const float* ctx       = (const float*)d_in[1];
  const float* emb_W     = (const float*)d_in[2];
  const float* ctx2ctx_W = (const float*)d_in[3];
  const float* hid2ctx_W = (const float*)d_in[4];
  const float* mlp_W     = (const float*)d_in[5];
  const float* mlp_b     = (const float*)d_in[6];
  const float* gru_Wih   = (const float*)d_in[7];
  const float* gru_Whh   = (const float*)d_in[8];
  const float* gru_bih   = (const float*)d_in[9];
  const float* gru_bhh   = (const float*)d_in[10];
  const float* hid2out_W = (const float*)d_in[11];
  const float* hid2out_b = (const float*)d_in[12];
  const float* ffctx_W   = (const float*)d_in[13];
  const float* ffctx_b   = (const float*)d_in[14];
  const float* out2prob_W= (const float*)d_in[15];
  const float* out2prob_b= (const float*)d_in[16];

  float* out = (float*)d_out;

  // ---- workspace layout: f32 region then bf16 region (~206 MB total) ----
  float* wsf = (float*)d_ws;
  float* ctx_p    = wsf;  wsf += (long)S_ * B_ * C_;   // 25,690,112
  float* y_emb    = wsf;  wsf += (long)T_ * B_ * E_;   //  4,194,304
  float* hproj    = wsf;  wsf += (long)B_ * C_;
  float* scores   = wsf;  wsf += (long)S_ * B_;
  float* alpha    = wsf;  wsf += (long)S_ * B_;
  float* gi       = wsf;  wsf += (long)B_ * G3_;
  float* gh       = wsf;  wsf += (long)B_ * G3_;
  float* hA       = wsf;  wsf += (long)B_ * H_;
  float* hB       = wsf;  wsf += (long)B_ * H_;
  float* logit    = wsf;  wsf += (long)B_ * E_;
  float* logits2  = wsf;  wsf += (long)B_ * V_;        //  2,560,000
  float* loss_part= wsf;  wsf += (long)(T_ - 1) * B_ + 64; // pad for alignment

  __bf16* wsb = (__bf16*)wsf;
  __bf16* ctx_bf    = wsb;  wsb += (long)S_ * B_ * C_; // 25,690,112 (51.4 MB)
  __bf16* xcat_bf   = wsb;  wsb += (long)B_ * 1024;
  __bf16* hA_bf     = wsb;  wsb += (long)B_ * H_;
  __bf16* hB_bf     = wsb;  wsb += (long)B_ * H_;
  __bf16* logit_bf  = wsb;  wsb += (long)B_ * E_;
  __bf16* p_ctx2ctx = wsb;  wsb += (long)C_ * C_;
  __bf16* p_hid2ctx = wsb;  wsb += (long)H_ * C_;
  __bf16* p_Wih     = wsb;  wsb += (long)1024 * G3_;
  __bf16* p_Whh     = wsb;  wsb += (long)H_ * G3_;
  __bf16* p_hid2out = wsb;  wsb += (long)H_ * E_;
  __bf16* p_ffctx   = wsb;  wsb += (long)C_ * E_;
  __bf16* p_out2prob= wsb;  wsb += (long)E_ * V_;      // 5,120,000

  dim3 blk256(256);

  // ---- one-time: weight packs, ctx bf16 mirror, h0, embeddings, ctx proj ----
  pack_b_kernel<<<dim3((C_ * C_)   / 256), blk256, 0, stream>>>(ctx2ctx_W, C_,  C_,   p_ctx2ctx);
  pack_b_kernel<<<dim3((H_ * C_)   / 256), blk256, 0, stream>>>(hid2ctx_W, C_,  H_,   p_hid2ctx);
  pack_b_kernel<<<dim3((1024 * G3_)/ 256), blk256, 0, stream>>>(gru_Wih,   G3_, 1024, p_Wih);
  pack_b_kernel<<<dim3((H_ * G3_)  / 256), blk256, 0, stream>>>(gru_Whh,   G3_, H_,   p_Whh);
  pack_b_kernel<<<dim3((H_ * E_)   / 256), blk256, 0, stream>>>(hid2out_W, E_,  H_,   p_hid2out);
  pack_b_kernel<<<dim3((C_ * E_)   / 256), blk256, 0, stream>>>(ffctx_W,   E_,  C_,   p_ffctx);
  pack_b_kernel<<<dim3((E_ * V_)   / 256), blk256, 0, stream>>>(out2prob_W,V_,  E_,   p_out2prob);

  {
    long n4 = ((long)S_ * B_ * C_) / 4;                // 6,422,528
    cvt_bf16_kernel<<<dim3((unsigned)((n4 + 255) / 256)), blk256, 0, stream>>>(
        ctx, ctx_bf, n4);
  }

  zero_h_kernel<<<dim3((B_ * H_ + 255) / 256), blk256, 0, stream>>>(hA, hA_bf, B_ * H_);
  embed_kernel<<<dim3((T_ * B_ * E_) / 256), blk256, 0, stream>>>(y, emb_W, y_emb);

  // ctx_p = ctx @ ctx2ctx_W   (M = S*B = 50176; clean bf16 path, 64-row strips)
  wmma_gemm_kernel<<<dim3((C_ / 16) / 4, (S_ * B_) / 64), dim3(128), 0, stream>>>(
      ctx_bf, C_, p_ctx2ctx, ctx_p, C_, S_ * B_, C_, C_, nullptr, 0);

  for (int t = 0; t < T_ - 1; ++t) {
    float*  h_cur    = (t & 1) ? hB    : hA;
    float*  h_nxt    = (t & 1) ? hA    : hB;
    __bf16* h_cur_bf = (t & 1) ? hB_bf : hA_bf;
    __bf16* h_nxt_bf = (t & 1) ? hA_bf : hB_bf;

    // hproj = h @ hid2ctx_W
    wmma_gemm_kernel<<<dim3((C_ / 16) / 4, B_ / 64), dim3(128), 0, stream>>>(
        h_cur_bf, H_, p_hid2ctx, hproj, C_, B_, C_, H_, nullptr, 0);

    // attention
    scores_kernel<<<dim3(S_ * B_), blk256, 0, stream>>>(ctx_p, hproj, mlp_W, mlp_b, scores);
    softmax_alpha_kernel<<<dim3(B_), blk256, 0, stream>>>(scores, alpha);
    zxcat_kernel<<<dim3((B_ * C_) / 256), blk256, 0, stream>>>(
        alpha, ctx, y_emb + (long)t * B_ * E_, xcat_bf);

    // GRU gates
    wmma_gemm_kernel<<<dim3((G3_ / 16) / 4, B_ / 64), dim3(128), 0, stream>>>(
        xcat_bf, 1024, p_Wih, gi, G3_, B_, G3_, 1024, gru_bih, 0);
    wmma_gemm_kernel<<<dim3((G3_ / 16) / 4, B_ / 64), dim3(128), 0, stream>>>(
        h_cur_bf, H_, p_Whh, gh, G3_, B_, G3_, H_, gru_bhh, 0);
    gru_kernel<<<dim3((B_ * H_) / 256), blk256, 0, stream>>>(gi, gh, h_cur, h_nxt, h_nxt_bf);

    // readout: logit = tanh(h_new@hid2out_W + b + z@ffctx_W + b + y_emb_t)
    wmma_gemm_kernel<<<dim3((E_ / 16) / 4, B_ / 64), dim3(128), 0, stream>>>(
        h_nxt_bf, H_, p_hid2out, logit, E_, B_, E_, H_, hid2out_b, 0);
    wmma_gemm_kernel<<<dim3((E_ / 16) / 4, B_ / 64), dim3(128), 0, stream>>>(
        xcat_bf + 512, 1024, p_ffctx, logit, E_, B_, E_, C_, ffctx_b, 1 /*accum*/);
    logit_finish_kernel<<<dim3((B_ * E_) / 256), blk256, 0, stream>>>(
        logit, y_emb + (long)t * B_ * E_, logit_bf);

    // vocab projection 256 x 10000 (625 n-tiles; grid guard covers remainder)
    wmma_gemm_kernel<<<dim3((V_ / 16 + 3) / 4, B_ / 64), dim3(128), 0, stream>>>(
        logit_bf, E_, p_out2prob, logits2, V_, B_, V_, E_, out2prob_b, 0);

    // log-softmax + masked NLL; logps stream straight to d_out[1:]
    logsoftmax_loss_kernel<<<dim3(B_), blk256, 0, stream>>>(
        logits2, y, t, out + 1, loss_part);
  }

  final_loss_kernel<<<dim3(1), blk256, 0, stream>>>(loss_part, out);
}
